// _MultiHead_44092134261077
// MI455X (gfx1250) — matrix operand order repaired and checked by reading, not compiled, not run on hardware
//
#include <hip/hip_runtime.h>

// ---------------------------------------------------------------------------
// MI455X (gfx1250) multi-head attention, bf16 WMMA + TDM path.
// d_out = [ out fp32 (4*2048*1024) | raw fp32 (4*16*2048*2048) ]
// d_ws  = [ Qh | Kh | Vt | Ctx ] bf16, each 4*16*2048*64 elements (16 MB)
//   Qh,Kh,Ctx layout: [b][h][n][64]   Vt layout: [b][h][64][n] (transposed)
// ---------------------------------------------------------------------------

typedef __attribute__((ext_vector_type(16))) __bf16 v16bf;
typedef __attribute__((ext_vector_type(8)))  float  v8f;
typedef unsigned int u32x4 __attribute__((ext_vector_type(4)));
typedef int          i32x8 __attribute__((ext_vector_type(8)));
typedef int          i32x4 __attribute__((ext_vector_type(4)));

union Frag16 { v16bf v; __bf16 e[16]; uint4 q[2]; };
union Frag8f { v8f   v; float  e[8];  };
union H4     { uint2 u; __bf16 h[4];  };

#define NHEADS   16
#define BATCH    4
#define SEQ      2048
#define DMODEL   1024
#define BHTOT    (BATCH * NHEADS)   // 64
#define MROWS    (BATCH * SEQ)      // 8192
#define SM_SCALE 0.125f             // 64^-0.5

static __device__ __forceinline__ __bf16 f2bf(float x) { return (__bf16)x; }

static __device__ __forceinline__ v8f wmma_bf(const Frag16& a, const Frag16& b, v8f c) {
  return __builtin_amdgcn_wmma_f32_16x16x32_bf16(false, a.v, false, b.v,
                                                 (short)0, c, false, false);
}

// ---------------------------------------------------------------------------
// TDM: 2D tile (data_size = 2B) global -> LDS with LDS row padding.
// pad_icode: pad after 2^(pad_icode+1) dwords; pad_acode: insert pad_acode+1 dwords.
// Descriptor bit layout per CDNA5 ISA 8.3/8.4 (group0 128b, group1 256b).
// Toolchain uses the 6-arg builtin: (g0, g1, g2, g3, g4, cpol).
// ---------------------------------------------------------------------------
static __device__ __forceinline__ void tdm_load_2d(
    unsigned lds_byte, const void* gptr,
    unsigned tile_d0, unsigned tile_d1,
    unsigned tensor_d0, unsigned tensor_d1,
    unsigned long long stride_d0_elems,
    unsigned pad_icode, unsigned pad_acode) {
  unsigned long long ga = (unsigned long long)gptr;
  u32x4 g0;
  g0[0] = 1u;                                   // count=1, user-mode descriptor
  g0[1] = lds_byte;                             // lds_addr (bytes)
  g0[2] = (unsigned)(ga & 0xffffffffu);         // global_addr[31:0]
  g0[3] = (unsigned)((ga >> 32) & 0x1ffffffu)   // global_addr[56:32]
          | (2u << 30);                         // type = 2 ("image")
  i32x8 g1;
  g1[0] = (int)((1u << 16)                      // data_size = 1 -> 2 bytes
          | (1u << 20)                          // pad_enable
          | (pad_icode << 22) | (pad_acode << 25));
  g1[1] = (int)((tensor_d0 & 0xffffu) << 16);                        // dim0[15:0] @ bits63:48
  g1[2] = (int)(((tensor_d0 >> 16) & 0xffffu) | ((tensor_d1 & 0xffffu) << 16));
  g1[3] = (int)(((tensor_d1 >> 16) & 0xffffu) | ((tile_d0 & 0xffffu) << 16));
  g1[4] = (int)(tile_d1 & 0xffffu);                                  // tile_dim1; tile_dim2=0
  g1[5] = (int)(stride_d0_elems & 0xffffffffu);
  g1[6] = (int)((stride_d0_elems >> 32) & 0xffffu);                  // stride hi; dim1_stride lo16=0
  g1[7] = 0;
  i32x4 z4 = {0, 0, 0, 0};
  i32x8 z8 = {0, 0, 0, 0, 0, 0, 0, 0};
  __builtin_amdgcn_tensor_load_to_lds(g0, g1, z4, z4, z8, 0);
}

// A-matrix fragment (bf16 16x32) from K-contiguous row: two 16B runs at
// K = 8*hf and K = 16 + 8*hf  (ISA 7.12.2).
static __device__ __forceinline__ void load_afrag(Frag16& f, const __bf16* row, int hf) {
  f.q[0] = *(const uint4*)(row + 8 * hf);
  f.q[1] = *(const uint4*)(row + 16 + 8 * hf);
}
// B-matrix fragment (bf16 32x16) from K-contiguous column row: two 16B runs at
// K = 8*hf and K = 16 + 8*hf (same order as A).
static __device__ __forceinline__ void load_bfrag(Frag16& f, const __bf16* col, int hf) {
  f.q[0] = *(const uint4*)(col + 8 * hf);
  f.q[1] = *(const uint4*)(col + 16 + 8 * hf);
}

// ---------------------------------------------------------------------------
// Projection GEMM: Y = X[8192,1024] @ W[1024,1024] + bias -> bf16 head-split.
// vmode=0: Yh[b][h][n][64];  vmode=1 (for V): Yh[b][h][64][n] transposed.
// Block 128 thr (4 waves), tile 64x64, BK=32.
// ---------------------------------------------------------------------------
#define ASTR  40   // 32 + 8 pad (row bytes = 80, 16B aligned)
#define BTSTR 40

__global__ __launch_bounds__(128)
void mha_proj_kernel(const float* __restrict__ X, const float* __restrict__ W,
                     const float* __restrict__ bias, __bf16* __restrict__ Yh,
                     int vmode) {
  __shared__ __align__(16) __bf16 As[64 * ASTR];   // [m][k]
  __shared__ __align__(16) __bf16 Bt[64 * BTSTR];  // [n][k] (transposed)
  const int tid  = threadIdx.x;
  const int wave = tid >> 5;
  const int lane = tid & 31;
  const int hf   = lane >> 4;
  const int nl   = lane & 15;
  const int m0 = blockIdx.x * 64;
  const int n0 = blockIdx.y * 64;

  Frag8f acc[4];
#pragma unroll
  for (int t = 0; t < 4; ++t)
#pragma unroll
    for (int i = 0; i < 8; ++i) acc[t].e[i] = 0.0f;

  for (int k0 = 0; k0 < DMODEL; k0 += 32) {
    __syncthreads();
    for (int i = tid * 4; i < 64 * 32; i += 128 * 4) {     // X tile, f32x4 -> bf16x4
      int r = i >> 5, c = i & 31;
      float4 f = *(const float4*)&X[(size_t)(m0 + r) * DMODEL + k0 + c];
      H4 h; h.h[0] = f2bf(f.x); h.h[1] = f2bf(f.y); h.h[2] = f2bf(f.z); h.h[3] = f2bf(f.w);
      *(uint2*)&As[r * ASTR + c] = h.u;
    }
    for (int i = tid; i < 32 * 64; i += 128) {             // W tile -> transposed bf16
      int r = i >> 6, c = i & 63;
      Bt[c * BTSTR + r] = f2bf(W[(size_t)(k0 + r) * DMODEL + n0 + c]);
    }
    __syncthreads();

    Frag16 a;
    load_afrag(a, &As[(wave * 16 + nl) * ASTR], hf);
#pragma unroll
    for (int t = 0; t < 4; ++t) {
      Frag16 b;
      load_bfrag(b, &Bt[(t * 16 + nl) * BTSTR], hf);
      acc[t].v = wmma_bf(a, b, acc[t].v);
    }
  }

#pragma unroll
  for (int t = 0; t < 4; ++t) {
    int C = n0 + t * 16 + nl;
    float bb = bias[C];
    int h = C >> 6, dh = C & 63;
#pragma unroll
    for (int i = 0; i < 8; ++i) {
      int R  = m0 + wave * 16 + i + 8 * hf;
      int bi = R >> 11, ni = R & 2047;
      size_t idx = vmode
        ? ((size_t)((bi * NHEADS + h) * 64 + dh)) * SEQ + ni      // [b][h][64][n]
        : ((size_t)((bi * NHEADS + h) * SEQ + ni)) * 64 + dh;     // [b][h][n][64]
      Yh[idx] = f2bf(acc[t].e[i] + bb);
    }
  }
}

// ---------------------------------------------------------------------------
// Output GEMM: out = Ctx(bf16 head-split) @ Wo + bo -> fp32 row-major.
// A tile fetched by TDM (bf16 straight copy), W transposed by ALL threads.
// ---------------------------------------------------------------------------
__global__ __launch_bounds__(128)
void mha_outproj_kernel(const __bf16* __restrict__ Ctx, const float* __restrict__ W,
                        const float* __restrict__ bias, float* __restrict__ Y) {
  __shared__ __align__(16) __bf16 As[64 * ASTR];
  __shared__ __align__(16) __bf16 Bt[64 * BTSTR];
  const int tid  = threadIdx.x;
  const int wave = tid >> 5;
  const int lane = tid & 31;
  const int hf   = lane >> 4;
  const int nl   = lane & 15;
  const int m0 = blockIdx.x * 64;
  const int n0 = blockIdx.y * 64;
  const unsigned ldsA = (unsigned)(size_t)(void*)As;
  const int bi = m0 >> 11, ni0 = m0 & 2047;

  Frag8f acc[4];
#pragma unroll
  for (int t = 0; t < 4; ++t)
#pragma unroll
    for (int i = 0; i < 8; ++i) acc[t].e[i] = 0.0f;

  for (int k0 = 0; k0 < DMODEL; k0 += 32) {
    __syncthreads();
    if (wave == 0) {
      // Ctx rows m0..m0+63, cols k0..k0+31 are contiguous within one head block
      int h = k0 >> 6, dh0 = k0 & 63;
      const __bf16* src = &Ctx[((size_t)((bi * NHEADS + h) * SEQ + ni0)) * 64 + dh0];
      // 32-elem rows (64B = 8x8B) -> pad code 3; pad 4 dwords -> code 3
      tdm_load_2d(ldsA, src, 32, 64, 32, 2048, 64, 3, 3);
    }
    for (int i = tid; i < 32 * 64; i += 128) {
      int r = i >> 6, c = i & 63;
      Bt[c * BTSTR + r] = f2bf(W[(size_t)(k0 + r) * DMODEL + n0 + c]);
    }
    if (wave == 0) __builtin_amdgcn_s_wait_tensorcnt(0);
    __syncthreads();

    Frag16 a;
    load_afrag(a, &As[(wave * 16 + nl) * ASTR], hf);
#pragma unroll
    for (int t = 0; t < 4; ++t) {
      Frag16 b;
      load_bfrag(b, &Bt[(t * 16 + nl) * BTSTR], hf);
      acc[t].v = wmma_bf(a, b, acc[t].v);
    }
  }

#pragma unroll
  for (int t = 0; t < 4; ++t) {
    int C = n0 + t * 16 + nl;
    float bb = bias[C];
#pragma unroll
    for (int i = 0; i < 8; ++i) {
      int R = m0 + wave * 16 + i + 8 * hf;
      Y[(size_t)R * DMODEL + C] = acc[t].e[i] + bb;
    }
  }
}

// ---------------------------------------------------------------------------
// Flash attention with raw-score side output.
// Grid (b*h, n/64); 128 thr (4 waves), wave owns 16 query rows.
// K/V tiles arrive via TDM; S = Q K^T (WMMA) -> scaled -> NT stores to raw;
// online softmax; P staged in LDS -> A frags; O += P @ V (WMMA).
// ---------------------------------------------------------------------------
#define KSTR 72
#define VSTR 72
#define PSTR 72

__global__ __launch_bounds__(128)
void mha_attn_kernel(const __bf16* __restrict__ Qh, const __bf16* __restrict__ Kh,
                     const __bf16* __restrict__ Vt, __bf16* __restrict__ Ctx,
                     float* __restrict__ raw) {
  __shared__ __align__(16) __bf16 Ks[64 * KSTR];   // [key][d]
  __shared__ __align__(16) __bf16 Vs[64 * VSTR];   // [d][key]
  __shared__ __align__(16) __bf16 Ps[64 * PSTR];   // per-wave P staging

  const int tid  = threadIdx.x;
  const int wave = tid >> 5;
  const int lane = tid & 31;
  const int hf   = lane >> 4;
  const int nl   = lane & 15;
  const int bh = blockIdx.x;
  const int qw = blockIdx.y * 64 + wave * 16;
  const unsigned ldsK = (unsigned)(size_t)(void*)Ks;
  const unsigned ldsV = (unsigned)(size_t)(void*)Vs;

  // Q fragments (16x64): 4x global_load_b128 per frag pair
  const __bf16* qp = Qh + ((size_t)bh * SEQ + qw) * 64 + nl * 64;
  Frag16 aq0, aq1;
  load_afrag(aq0, qp, hf);
  load_afrag(aq1, qp + 32, hf);

  Frag8f acc[4];
  float mrow[8], lrow[8];
#pragma unroll
  for (int t = 0; t < 4; ++t)
#pragma unroll
    for (int i = 0; i < 8; ++i) acc[t].e[i] = 0.0f;
#pragma unroll
  for (int i = 0; i < 8; ++i) { mrow[i] = -3.0e38f; lrow[i] = 0.0f; }

  float*  rawq = raw + ((size_t)bh * SEQ + qw) * SEQ;
  __bf16* Pw   = Ps + wave * 16 * PSTR;
  const __bf16* Khead = Kh + (size_t)bh * SEQ * 64;   // [n][64]
  const __bf16* Vhead = Vt + (size_t)bh * 64 * SEQ;   // [64][n]

  for (int kt = 0; kt < 32; ++kt) {
    __syncthreads();
    if (wave == 0) {
      // K tile: rows=64 keys, 64 dims contiguous; 64-elem rows -> pad codes (4,3)
      tdm_load_2d(ldsK, Khead + (size_t)kt * 64 * 64, 64, 64, 64, 2048, 64, 4, 3);
      // V tile: rows=64 dims, keys contiguous with stride SEQ
      tdm_load_2d(ldsV, Vhead + (size_t)kt * 64, 64, 64, 2048, 64, 2048, 4, 3);
      __builtin_amdgcn_s_wait_tensorcnt(0);
    }
    __syncthreads();

    // ---- S = Q K^T (4 subtiles of 16 keys), scale, stream to raw ----
    Frag8f s[4];
#pragma unroll
    for (int jt = 0; jt < 4; ++jt) {
      const __bf16* krow = &Ks[(jt * 16 + nl) * KSTR];
      Frag16 b0, b1;
      load_bfrag(b0, krow, hf);
      load_bfrag(b1, krow + 32, hf);
      Frag8f z;
#pragma unroll
      for (int i = 0; i < 8; ++i) z.e[i] = 0.0f;
      s[jt].v = wmma_bf(aq0, b0, z.v);
      s[jt].v = wmma_bf(aq1, b1, s[jt].v);
#pragma unroll
      for (int i = 0; i < 8; ++i) s[jt].e[i] *= SM_SCALE;
      int kc = kt * 64 + jt * 16 + nl;
#pragma unroll
      for (int i = 0; i < 8; ++i) {
        int m = i + 8 * hf;
        __builtin_nontemporal_store(s[jt].e[i], &rawq[(size_t)m * SEQ + kc]);
      }
    }

    // ---- online softmax (reduce across 16-lane column groups) ----
    float tmax[8];
#pragma unroll
    for (int i = 0; i < 8; ++i)
      tmax[i] = fmaxf(fmaxf(s[0].e[i], s[1].e[i]), fmaxf(s[2].e[i], s[3].e[i]));
#pragma unroll
    for (int off = 8; off >= 1; off >>= 1)
#pragma unroll
      for (int i = 0; i < 8; ++i)
        tmax[i] = fmaxf(tmax[i], __shfl_xor(tmax[i], off, 32));

    float corr[8];
#pragma unroll
    for (int i = 0; i < 8; ++i) {
      float mn = fmaxf(mrow[i], tmax[i]);
      corr[i]  = __expf(mrow[i] - mn);
      mrow[i]  = mn;
    }
    float rsum[8];
#pragma unroll
    for (int i = 0; i < 8; ++i) {
      float rs = 0.0f;
#pragma unroll
      for (int jt = 0; jt < 4; ++jt) {
        float pv = __expf(s[jt].e[i] - mrow[i]);
        s[jt].e[i] = pv;            // reuse s as P
        rs += pv;
      }
      rsum[i] = rs;
    }
#pragma unroll
    for (int off = 8; off >= 1; off >>= 1)
#pragma unroll
      for (int i = 0; i < 8; ++i)
        rsum[i] += __shfl_xor(rsum[i], off, 32);
#pragma unroll
    for (int i = 0; i < 8; ++i) lrow[i] = lrow[i] * corr[i] + rsum[i];
#pragma unroll
    for (int t = 0; t < 4; ++t)
#pragma unroll
      for (int i = 0; i < 8; ++i) acc[t].e[i] *= corr[i];

    // ---- P (D layout) -> LDS; O += P @ V ----
#pragma unroll
    for (int jt = 0; jt < 4; ++jt)
#pragma unroll
      for (int i = 0; i < 8; ++i)
        Pw[(i + 8 * hf) * PSTR + jt * 16 + nl] = f2bf(s[jt].e[i]);

#pragma unroll
    for (int kc = 0; kc < 2; ++kc) {
      Frag16 ap;
      load_afrag(ap, &Pw[nl * PSTR + kc * 32], hf);
#pragma unroll
      for (int t = 0; t < 4; ++t) {
        Frag16 bv;
        load_bfrag(bv, &Vs[(t * 16 + nl) * VSTR + kc * 32], hf);
        acc[t].v = wmma_bf(ap, bv, acc[t].v);
      }
    }
  }

  // ---- normalize + store context (bf16, head-split) ----
  __bf16* Cg = Ctx + ((size_t)bh * SEQ + qw) * 64;
  float inv[8];
#pragma unroll
  for (int i = 0; i < 8; ++i) inv[i] = 1.0f / lrow[i];
#pragma unroll
  for (int t = 0; t < 4; ++t)
#pragma unroll
    for (int i = 0; i < 8; ++i) {
      int m = i + 8 * hf;
      Cg[(size_t)m * 64 + t * 16 + nl] = f2bf(acc[t].e[i] * inv[i]);
    }
}

// ---------------------------------------------------------------------------
extern "C" void kernel_launch(void* const* d_in, const int* in_sizes, int n_in,
                              void* d_out, int out_size, void* d_ws, size_t ws_size,
                              hipStream_t stream) {
  (void)in_sizes; (void)n_in; (void)out_size; (void)ws_size;
  const float* q  = (const float*)d_in[0];
  const float* k  = (const float*)d_in[1];
  const float* v  = (const float*)d_in[2];
  const float* Wq = (const float*)d_in[3];
  const float* bq = (const float*)d_in[4];
  const float* Wk = (const float*)d_in[5];
  const float* bk = (const float*)d_in[6];
  const float* Wv = (const float*)d_in[7];
  const float* bv = (const float*)d_in[8];
  const float* Wo = (const float*)d_in[9];
  const float* bo = (const float*)d_in[10];

  float* out = (float*)d_out;                          // [4,2048,1024]
  float* raw = out + (size_t)BATCH * SEQ * DMODEL;     // [4,16,2048,2048]

  const size_t HSZ = (size_t)BHTOT * SEQ * 64;
  __bf16* Qh  = (__bf16*)d_ws;
  __bf16* Kh  = Qh + HSZ;
  __bf16* Vt  = Kh + HSZ;
  __bf16* Ctx = Vt + HSZ;

  dim3 gproj(MROWS / 64, DMODEL / 64);                 // 128 x 16
  mha_proj_kernel<<<gproj, 128, 0, stream>>>(q, Wq, bq, Qh, 0);
  mha_proj_kernel<<<gproj, 128, 0, stream>>>(k, Wk, bk, Kh, 0);
  mha_proj_kernel<<<gproj, 128, 0, stream>>>(v, Wv, bv, Vt, 1);

  mha_attn_kernel<<<dim3(BHTOT, SEQ / 64), 128, 0, stream>>>(Qh, Kh, Vt, Ctx, raw);

  mha_outproj_kernel<<<gproj, 128, 0, stream>>>(Ctx, Wo, bo, out);
}
